// LSTMConv2DBlock_58042188038304
// MI455X (gfx1250) — matrix operand order repaired and checked by reading, not compiled, not run on hardware
//
#include <hip/hip_runtime.h>
#include <hip/hip_bf16.h>
#include <stdint.h>

// ---------------------------------------------------------------------------
// x: (B=4, T=16, H=16, W=21, Cin=256); F=256; 4F=1024; 5x5 SAME.
// H*W = 336 = 21 tiles of 16 px. Padded images: (H+4)x(W+4) = 20x25 = 500 px.
// ---------------------------------------------------------------------------
#define BB 4
#define TT 16
#define HH 16
#define WW 21
#define CIN 256
#define FF 256
#define F4 1024
#define HW 336
#define PH 20
#define PW 25
#define PHW 500
#define PIMG (PHW * CIN)     // 128,000 elements per padded image
#define MT_IMG 21            // M-tiles per image
#define NT 64                // N-tiles (1024/16)
#define NTW 4                // N-tiles per wave
#define KB 8                 // K-blocks of 32 over Cin=256
#define TAPS 25

typedef __attribute__((ext_vector_type(16))) __bf16 v16bf;
typedef __attribute__((ext_vector_type(8)))  __bf16 v8bf;
typedef __attribute__((ext_vector_type(8)))  float  v8f;

__device__ __forceinline__ float hsig(float x) {
  return fminf(fmaxf(0.2f * x + 0.5f, 0.0f), 1.0f);
}

__device__ __forceinline__ v16bf combine(v8bf lo, v8bf hi) {
  union { v16bf v; v8bf h[2]; } u;
  u.h[0] = lo; u.h[1] = hi;
  return u.v;
}

// ---------------------------------------------------------------------------
// Weight repack: (5,5,256,1024) fp32 HWIO -> bf16 B-fragment layout:
//   packed[(((tap*64 + nt)*8 + kblk)*512) + lane*16 + e]
//   lane l: n = l&15, hi = l>>4, K = kblk*32 + hi*16 + e   (contiguous k order, same as A)
// ---------------------------------------------------------------------------
__global__ __launch_bounds__(256) void pack_w(const float* __restrict__ w,
                                              __bf16* __restrict__ wp) {
  int idx = blockIdx.x * 256 + threadIdx.x;          // < 25*64*8*512 = 6,553,600
  int e    =  idx        & 15;
  int l    = (idx >> 4)  & 31;
  int kblk = (idx >> 9)  & 7;
  int nt   = (idx >> 12) & 63;
  int tap  =  idx >> 18;
  int n  = l & 15;
  int hi = l >> 4;
  int K  = kblk * 32 + hi * 16 + e;
  wp[idx] = (__bf16)w[((size_t)(tap * CIN + K)) * F4 + nt * 16 + n];
}

// Zero an entire bf16 buffer (padded-activation halo init).
__global__ __launch_bounds__(256) void zero_bf(__bf16* __restrict__ y, int n) {
  int idx = blockIdx.x * 256 + threadIdx.x;
  if (idx < n) y[idx] = (__bf16)0.0f;
}

// fp32 (nImg,336,256) -> bf16 padded (nImg,20,25,256) interior
__global__ __launch_bounds__(256) void f2bf_pad(const float* __restrict__ x,
                                                __bf16* __restrict__ y) {
  int idx = blockIdx.x * 256 + threadIdx.x;          // < nImg*336*256
  int img = idx / (HW * CIN);
  int r   = idx % (HW * CIN);
  int p   = r >> 8;
  int ch  = r & 255;
  int h = p / WW, w = p % WW;
  y[(size_t)img * PIMG + ((h + 2) * PW + (w + 2)) * CIN + ch] = (__bf16)x[idx];
}

// zero cell state (fp32) and the full padded h buffer (bf16)
__global__ __launch_bounds__(256) void zero_state(float* __restrict__ c,
                                                  __bf16* __restrict__ hp) {
  int idx = blockIdx.x * 256 + threadIdx.x;          // < BB*PIMG = 512,000
  if (idx < BB * HW * FF) c[idx] = 0.0f;
  if (idx < BB * PIMG)    hp[idx] = (__bf16)0.0f;
}

// ---------------------------------------------------------------------------
// Implicit-GEMM 5x5 conv (Cin=256 -> Cout=1024), bf16 WMMA, f32 accum.
// One wave32: 16 output pixels x 4 N-tiles (64 Cout) -> 800 WMMAs per wave.
// Halo-padded activations -> branch-free K-loop, EXEC stays all-ones.
// Per kb-step: A loads + ALL 4 B-tile loads issue as one clause, then the
// 4 WMMAs; launch_bounds(256,1) relaxes the register budget so the scheduler
// can keep the next step's fragments in flight (wait slack instead of drain).
//   actp   : (nImg, 20, 25, 256) bf16, zero halo
//   wp     : packed weights
//   bias   : (1024) or nullptr      (accumulator init)
//   accum  : (nImgOut,336,1024) or nullptr (fused add init)
//   out    : (nImgOut,336,1024) fp32
//   transposeOut: img=b*16+t -> oimg=t*4+b
// ---------------------------------------------------------------------------
__global__ __launch_bounds__(256, 1) void conv5x5_wmma(
    const __bf16* __restrict__ actp,
    const __bf16* __restrict__ wp,
    const float*  __restrict__ bias,
    const float*  __restrict__ accum,
    float* __restrict__ out,
    int nImg, int transposeOut) {
  int wid   = blockIdx.x * 8 + (threadIdx.x >> 5);
  int total = nImg * MT_IMG * (NT / NTW);
  if (wid >= total) return;                          // wave-uniform exit

  int mt  = wid >> 4;          // image*21 + ptile
  int ng  = wid & 15;          // N-group (4 tiles)
  int img = mt / MT_IMG;
  int pt  = mt % MT_IMG;

  int l  = threadIdx.x & 31;
  int m  = l & 15;
  int hi = l >> 4;
  int n  = m;

  int oimg = transposeOut ? ((img & 15) * BB + (img >> 4)) : img;

  int p = pt * 16 + m;
  int h = p / WW;
  int w = p % WW;

  // Per-lane A base at padded (h, w); tap (dy,dx) is a non-negative offset.
  const __bf16* abase = actp + (size_t)img * PIMG + (h * PW + w) * CIN + hi * 16;
  const int ntbase = ng * NTW;

  // ---- accumulator init ----
  v8f acc[NTW];
  if (accum) {
#pragma unroll
    for (int j = 0; j < NTW; ++j) {
      const float* ap = accum + ((size_t)oimg * HW + pt * 16) * F4 + (ntbase + j) * 16 + n;
#pragma unroll
      for (int i = 0; i < 8; ++i) acc[j][i] = ap[(size_t)(i + 8 * hi) * F4];
    }
  } else if (bias) {
#pragma unroll
    for (int j = 0; j < NTW; ++j) {
      float bv = bias[(ntbase + j) * 16 + n];
#pragma unroll
      for (int i = 0; i < 8; ++i) acc[j][i] = bv;
    }
  } else {
#pragma unroll
    for (int j = 0; j < NTW; ++j)
#pragma unroll
      for (int i = 0; i < 8; ++i) acc[j][i] = 0.0f;
  }

  // ---- K loop: 25 taps x 8 channel-blocks, branch-free ----
  const size_t WTAP = (size_t)NT * KB * 512;         // packed elements per tap
  const __bf16* wptr = wp + l * 16 + (size_t)ntbase * KB * 512;
  const __bf16* arow = abase;

#pragma unroll 1
  for (int dy = 0; dy < 5; ++dy) {
    const __bf16* aptr = arow;
#pragma unroll 1
    for (int dx = 0; dx < 5; ++dx) {
      __builtin_prefetch(wptr + WTAP, 0, 1);         // next tap's weights
#pragma unroll 4
      for (int kb = 0; kb < KB; ++kb) {
        // A fragment: elems 0..15 = K[hi*16..+16)
        v16bf Av = combine(
            *reinterpret_cast<const v8bf*>(aptr + kb * 32),
            *reinterpret_cast<const v8bf*>(aptr + kb * 32 + 8));
        // issue all 4 B-tile loads before any WMMA consumes them
        v16bf Bv[NTW];
#pragma unroll
        for (int j = 0; j < NTW; ++j) {
          const __bf16* wb = wptr + (size_t)j * KB * 512 + kb * 512;
          Bv[j] = combine(*reinterpret_cast<const v8bf*>(wb),
                          *reinterpret_cast<const v8bf*>(wb + 8));
        }
#pragma unroll
        for (int j = 0; j < NTW; ++j)
          acc[j] = __builtin_amdgcn_wmma_f32_16x16x32_bf16(
              false, Av, false, Bv[j], (short)0, acc[j], false, false);
      }
      aptr += CIN;      // dx+1
      wptr += WTAP;     // next tap
    }
    arow += PW * CIN;   // dy+1
  }

  // ---- store (C/D layout: VGPR i = row i + 8*hi, col n) ----
#pragma unroll
  for (int j = 0; j < NTW; ++j) {
    float* op = out + ((size_t)oimg * HW + pt * 16) * F4 + (ntbase + j) * 16 + n;
#pragma unroll
    for (int i = 0; i < 8; ++i) op[(size_t)(i + 8 * hi) * F4] = acc[j][i];
  }
}

// ---------------------------------------------------------------------------
// LSTM gates for one timestep.  g: (B,336,1024) fused xg[t]+conv(h,rk).
// Writes c (in place), seq[t] (fp32), padded bf16 h for next recurrent conv.
// ---------------------------------------------------------------------------
__global__ __launch_bounds__(256) void lstm_pointwise(
    const float* __restrict__ g, float* __restrict__ c,
    float* __restrict__ seq, __bf16* __restrict__ hp, int t) {
  int idx = blockIdx.x * 256 + threadIdx.x;          // < B*336*256
  int pix = idx >> 8;
  int ch  = idx & 255;
  const float* gp = g + (size_t)pix * F4 + ch;
  float fi = hsig(gp[0]);
  float ff = hsig(gp[FF]);
  float cc = fmaxf(gp[2 * FF], 0.0f);
  float fo = hsig(gp[3 * FF]);
  float cn = ff * c[idx] + fi * cc;
  float hh = fo * fmaxf(cn, 0.0f);
  c[idx] = cn;
  int b = pix / HW, pp = pix % HW;
  seq[((size_t)(b * TT + t) * HW + pp) * FF + ch] = hh;
  int y = pp / WW, x = pp % WW;
  hp[(size_t)b * PIMG + ((y + 2) * PW + (x + 2)) * CIN + ch] = (__bf16)hh;
}

// Instance-norm stats: one block per (b, channel); reduce over T*H*W = 5376.
__global__ __launch_bounds__(256) void inst_stats(const float* __restrict__ seq,
                                                  float* __restrict__ mu,
                                                  float* __restrict__ rs) {
  int bc = blockIdx.x;                               // < B*F = 1024
  int b  = bc >> 8;
  int ch = bc & 255;
  const float* base = seq + (size_t)b * TT * HW * FF + ch;
  float s = 0.0f, sq = 0.0f;
  for (int i = threadIdx.x; i < TT * HW; i += 256) {
    float v = base[(size_t)i * FF];
    s += v; sq += v * v;
  }
  __shared__ float ss[256], s2[256];
  ss[threadIdx.x] = s; s2[threadIdx.x] = sq;
  __syncthreads();
  for (int st = 128; st > 0; st >>= 1) {
    if (threadIdx.x < st) {
      ss[threadIdx.x] += ss[threadIdx.x + st];
      s2[threadIdx.x] += s2[threadIdx.x + st];
    }
    __syncthreads();
  }
  if (threadIdx.x == 0) {
    float inv = 1.0f / (float)(TT * HW);
    float mm  = ss[0] * inv;
    float var = s2[0] * inv - mm * mm;
    mu[bc] = mm;
    rs[bc] = rsqrtf(var + 1e-3f);
  }
}

// Apply instance norm; fp32 out (final) or bf16 padded activations (layer 1->2).
__global__ __launch_bounds__(256) void norm_apply(
    const float* __restrict__ seq, const float* __restrict__ mu,
    const float* __restrict__ rs, const float* __restrict__ gamma,
    const float* __restrict__ beta, float* __restrict__ dst_f,
    __bf16* __restrict__ dst_pad) {
  int idx = blockIdx.x * 256 + threadIdx.x;          // < B*T*336*256
  int ch = idx & 255;
  int b  = idx / (TT * HW * FF);
  float v = (seq[idx] - mu[b * FF + ch]) * rs[b * FF + ch] * gamma[ch] + beta[ch];
  if (dst_f) {
    dst_f[idx] = v;
  } else {
    int img = idx / (HW * FF);                       // b*16 + t
    int r   = idx % (HW * FF);
    int p   = r >> 8;
    int h = p / WW, w = p % WW;
    dst_pad[(size_t)img * PIMG + ((h + 2) * PW + (w + 2)) * CIN + ch] = (__bf16)v;
  }
}

// h2 = seq2[:, T-1] (pre-norm), c2 = cell state; append to d_out.
__global__ __launch_bounds__(256) void final_copy(const float* __restrict__ seq,
                                                  const float* __restrict__ c,
                                                  float* __restrict__ out_h2,
                                                  float* __restrict__ out_c2) {
  int idx = blockIdx.x * 256 + threadIdx.x;          // < B*336*256
  int b = idx / (HW * FF);
  int r = idx % (HW * FF);
  out_h2[idx] = seq[(size_t)(b * TT + (TT - 1)) * HW * FF + r];
  out_c2[idx] = c[idx];
}

// ---------------------------------------------------------------------------
// Orchestration
// ---------------------------------------------------------------------------
extern "C" void kernel_launch(void* const* d_in, const int* in_sizes, int n_in,
                              void* d_out, int out_size, void* d_ws, size_t ws_size,
                              hipStream_t stream) {
  (void)in_sizes; (void)n_in; (void)out_size; (void)ws_size;
  const float* x   = (const float*)d_in[0];
  const float* k1  = (const float*)d_in[1];
  const float* rk1 = (const float*)d_in[2];
  const float* b1  = (const float*)d_in[3];
  const float* g1  = (const float*)d_in[4];
  const float* bt1 = (const float*)d_in[5];
  const float* k2  = (const float*)d_in[6];
  const float* rk2 = (const float*)d_in[7];
  const float* b2  = (const float*)d_in[8];
  const float* g2  = (const float*)d_in[9];
  const float* bt2 = (const float*)d_in[10];
  float* out = (float*)d_out;

  const size_t N_XG    = (size_t)TT * BB * HW * F4;    // 22,020,096 (T,B,336,1024)
  const size_t N_SEQ   = (size_t)BB * TT * HW * FF;    //  5,505,024
  const size_t N_STATE = (size_t)BB * HW * FF;         //    344,064
  const size_t N_ACTP  = (size_t)BB * TT * PIMG;       //  8,192,000 padded bf16
  const size_t N_HP    = (size_t)BB * PIMG;            //    512,000 padded bf16
  const size_t N_WP    = (size_t)TAPS * NT * KB * 512; //  6,553,600

  char* ws = (char*)d_ws;
  size_t o = 0;
  auto carve = [&](size_t bytes) -> char* {
    char* p = ws + o;
    o = (o + bytes + 255) & ~(size_t)255;
    return p;
  };
  float*  xg    = (float*) carve(N_XG * 4);
  float*  seq   = (float*) carve(N_SEQ * 4);
  __bf16* actp  = (__bf16*)carve(N_ACTP * 2);
  __bf16* hp    = (__bf16*)carve(N_HP * 2);
  float*  cbuf  = (float*) carve(N_STATE * 4);
  __bf16* wpk1  = (__bf16*)carve(N_WP * 2);
  __bf16* wprk1 = (__bf16*)carve(N_WP * 2);
  __bf16* wpk2  = (__bf16*)carve(N_WP * 2);
  __bf16* wprk2 = (__bf16*)carve(N_WP * 2);
  float*  mu    = (float*) carve(BB * FF * 4);
  float*  rs    = (float*) carve(BB * FF * 4);

  // Data prep (deterministic every call)
  pack_w<<<25600, 256, 0, stream>>>(k1,  wpk1);
  pack_w<<<25600, 256, 0, stream>>>(rk1, wprk1);
  pack_w<<<25600, 256, 0, stream>>>(k2,  wpk2);
  pack_w<<<25600, 256, 0, stream>>>(rk2, wprk2);
  zero_bf<<<(int)((N_ACTP + 255) / 256), 256, 0, stream>>>(actp, (int)N_ACTP);
  f2bf_pad<<<21504, 256, 0, stream>>>(x, actp);

  for (int layer = 0; layer < 2; ++layer) {
    const __bf16* wpk  = layer ? wpk2  : wpk1;
    const __bf16* wpr  = layer ? wprk2 : wprk1;
    const float*  bias = layer ? b2 : b1;

    // xg = conv(act, k) + bias -> (T,B,336,1024): 64*21*16 = 21504 waves
    conv5x5_wmma<<<2688, 256, 0, stream>>>(actp, wpk, bias, nullptr, xg,
                                           BB * TT, /*transposeOut=*/1);
    zero_state<<<(int)((N_HP + 255) / 256), 256, 0, stream>>>(cbuf, hp);

    for (int t = 0; t < TT; ++t) {
      float* xgt = xg + (size_t)t * BB * HW * F4;
      // g = xg[t] + conv(h, rk): 4*21*16 = 1344 waves
      conv5x5_wmma<<<168, 256, 0, stream>>>(hp, wpr, nullptr, xgt, xgt,
                                            BB, /*transposeOut=*/0);
      lstm_pointwise<<<1344, 256, 0, stream>>>(xgt, cbuf, seq, hp, t);
    }

    inst_stats<<<BB * FF, 256, 0, stream>>>(seq, mu, rs);
    if (layer == 0)
      norm_apply<<<21504, 256, 0, stream>>>(seq, mu, rs, g1, bt1, nullptr, actp);
    else
      norm_apply<<<21504, 256, 0, stream>>>(seq, mu, rs, g2, bt2, out, nullptr);
  }

  final_copy<<<1344, 256, 0, stream>>>(seq, cbuf,
                                       out + N_SEQ, out + N_SEQ + N_STATE);
}
